// ActorCritic_89008902242317
// MI455X (gfx1250) — matrix operand order repaired and checked by reading, not compiled, not run on hardware
//
#include <hip/hip_runtime.h>
#include <hip/hip_bf16.h>

// ---------------- problem constants ----------------
#define KNOW 1024
#define ITEM 20000
#define DD   128
#define LL   2
#define LAT  256
#define BB   256
#define SS   200
#define NTOT (ITEM + KNOW)   // 21024
#define E1N  32768
#define E2N  160000

typedef __attribute__((ext_vector_type(16))) _Float16 v16h;
typedef __attribute__((ext_vector_type(8)))  _Float16 h8;
typedef __attribute__((ext_vector_type(4)))  _Float16 h4;
typedef __attribute__((ext_vector_type(8)))  float    v8f;
typedef __attribute__((ext_vector_type(4)))  float    f4v;
typedef __attribute__((ext_vector_type(4)))  unsigned int u32x4;
typedef __attribute__((ext_vector_type(4)))  int      i32x4;
typedef __attribute__((ext_vector_type(8)))  int      i32x8;

#if __has_builtin(__builtin_amdgcn_tensor_load_to_lds)
#define USE_TDM 1
#else
#define USE_TDM 0
#endif

#if USE_TDM
// Issue a TDM load of a 64-row x 64-col f32 tile of A (row stride K) into LDS.
// D# pad: +4 DWORDs every 64 -> LDS row stride 68 f32 (bank-conflict-free).
// tensor_dim0/1 set to the remaining extent so OOB rows/cols arrive as zeros.
__device__ __forceinline__ void tdm_issue_a(const float* A, unsigned lds_addr,
                                            int bm, int kb, int M, int K)
{
  unsigned long long ga =
      (unsigned long long)(const void*)(A + (size_t)bm * K + kb);
  unsigned td0 = (unsigned)(K - kb);
  unsigned td1 = (unsigned)(M - bm);
  u32x4 g0;
  g0[0] = 1u;                                                // count=1, user mode
  g0[1] = lds_addr;                                          // LDS byte address
  g0[2] = (unsigned)ga;                                      // global addr [31:0]
  g0[3] = (unsigned)((ga >> 32) & 0x01ffffffu) | (2u << 30); // [56:32] | type=2
  i32x8 g1;
  // data_size=4B | pad_enable | pad_interval=64dw | pad_amount=4dw
  g1[0] = (int)((2u << 16) | (1u << 20) | (5u << 22) | (3u << 25));
  g1[1] = (int)((td0 & 0xffffu) << 16);                      // tensor_dim0[15:0]
  g1[2] = (int)(((td0 >> 16) & 0xffffu) | ((td1 & 0xffffu) << 16));
  g1[3] = (int)(((td1 >> 16) & 0xffffu) | (64u << 16));      // tile_dim0=64
  g1[4] = 64;                                                // tile_dim1=64
  g1[5] = (int)(unsigned)K;                                  // dim0 stride low 32
  g1[6] = 0;
  g1[7] = 0;
  i32x4 gz = {0, 0, 0, 0};
#if defined(__clang_major__) && (__clang_major__ >= 23)
  i32x8 gz8 = {0, 0, 0, 0, 0, 0, 0, 0};
  __builtin_amdgcn_tensor_load_to_lds(g0, g1, gz, gz, gz8, 0);
#else
  __builtin_amdgcn_tensor_load_to_lds(g0, g1, gz, gz, 0);
#endif
}
#endif

// ============================================================
// Generic GEMM  C = act(A @ B + bias)    (fp32 in/out, f16 WMMA)
// A: MxK row major, B: KxN row major, C: MxN (ldc stride)
// Requires: K % 64 == 0.  M and N guarded.
// Block: 256 threads = 8 wave32; tile 64(M) x 64(N) x 64(K).
// Wave grid 4x2; each wave computes 16x32 (two fp32 accumulators,
// A fragment reused) -> 4 v_wmma_f32_16x16x32_f16 per K-stage.
// A tiles double-buffered via the Tensor Data Mover: next tile's DMA
// is issued before s_wait_tensorcnt(1) on the current one, overlapping
// the DMA with the whole compute stage.
// ============================================================
__global__ __launch_bounds__(256) void gemm_wmma_f16(
    const float* __restrict__ A, const float* __restrict__ Bm,
    const float* __restrict__ bias, float* __restrict__ C,
    int M, int N, int K, int ldc, int act)
{
#if USE_TDM
  __shared__ __align__(16) float AsF[2][64 * 68]; // TDM-padded rows (stride 68)
#else
  __shared__ __align__(16) _Float16 As[64][72];   // 64 rows x 64 k (+8 pad)
#endif
  __shared__ __align__(16) _Float16 BsT[64][72];  // 64 cols x 64 k (+8 pad)

  const int tid  = threadIdx.x;
  const int lane = tid & 31;
  const int wave = tid >> 5;       // 0..7
  const int wm   = wave & 3;       // 4 waves along M
  const int wn   = wave >> 2;      // 2 waves along N (32 cols each)
  const int bm   = blockIdx.y * 64;
  const int bn   = blockIdx.x * 64;
  const int lo   = lane & 15;
  const int hi   = lane >> 4;

  v8f acc0 = {0.f,0.f,0.f,0.f,0.f,0.f,0.f,0.f};
  v8f acc1 = {0.f,0.f,0.f,0.f,0.f,0.f,0.f,0.f};

#if USE_TDM
  if (wave == 0)
    tdm_issue_a(A, (unsigned)(size_t)&AsF[0][0], bm, 0, M, K);
#endif

  for (int kb = 0; kb < K; kb += 64) {
#if USE_TDM
    const int buf = (kb >> 6) & 1;
    if (wave == 0 && kb + 64 < K)     // prefetch next A tile into other buffer
      tdm_issue_a(A, (unsigned)(size_t)&AsF[buf ^ 1][0], bm, kb + 64, M, K);
#else
    // ---- fallback: stage A tile 64x64 via VALU (float4 -> h4) ----
#pragma unroll
    for (int it = 0; it < 4; ++it) {
      int idx = tid + it * 256;
      int r   = idx >> 4;
      int c4  = (idx & 15) << 2;
      int gr  = bm + r;
      h4 v = {(_Float16)0, (_Float16)0, (_Float16)0, (_Float16)0};
      if (gr < M) {
        f4v f = *(const f4v*)(A + (size_t)gr * K + kb + c4);
        v[0] = (_Float16)f.x; v[1] = (_Float16)f.y;
        v[2] = (_Float16)f.z; v[3] = (_Float16)f.w;
      }
      *(h4*)&As[r][c4] = v;
    }
#endif
    // ---- stage B tile 64(K) x 64(N) transposed, N-guarded ----
#pragma unroll
    for (int it = 0; it < 4; ++it) {
      int idx = tid + it * 256;        // 1024 slots (1 col x 4 k-rows each)
      int c   = idx & 63;
      int r4  = (idx >> 6) << 2;
      int gc  = bn + c;
      h4 v = {(_Float16)0, (_Float16)0, (_Float16)0, (_Float16)0};
      if (gc < N) {
#pragma unroll
        for (int u = 0; u < 4; ++u)
          v[u] = (_Float16)Bm[(size_t)(kb + r4 + u) * N + gc];
        if (kb + 64 < K)
          __builtin_prefetch(Bm + (size_t)(kb + 64 + r4) * N + gc, 0, 0);
      }
      *(h4*)&BsT[c][r4] = v;
    }
#if USE_TDM
    if (wave == 0) {
      if (kb + 64 < K) __builtin_amdgcn_s_wait_tensorcnt((short)1);
      else             __builtin_amdgcn_s_wait_tensorcnt((short)0);
    }
#endif
    __syncthreads();

    const int ar = wm * 16 + lo;
    const int bc = wn * 32 + lo;
#pragma unroll
    for (int kk = 0; kk < 64; kk += 32) {
      v16h af, bf0, bf1;
#if USE_TDM
      // A 16x32 frag from f32 LDS: j<8 -> k=kk+8*hi+j ; j>=8 -> +16
      const float* arow = &AsF[buf][(size_t)ar * 68];
      f4v a00 = *(const f4v*)(arow + kk + 8 * hi);
      f4v a01 = *(const f4v*)(arow + kk + 8 * hi + 4);
      f4v a10 = *(const f4v*)(arow + kk + 16 + 8 * hi);
      f4v a11 = *(const f4v*)(arow + kk + 16 + 8 * hi + 4);
#pragma unroll
      for (int j = 0; j < 4; ++j) {
        af[j]      = (_Float16)a00[j];
        af[j + 4]  = (_Float16)a01[j];
        af[j + 8]  = (_Float16)a10[j];
        af[j + 12] = (_Float16)a11[j];
      }
#else
      h8 a0 = *(const h8*)&As[ar][kk + 8 * hi];
      h8 a1 = *(const h8*)&As[ar][kk + 16 + 8 * hi];
      af = __builtin_shufflevector(a0, a1, 0,1,2,3,4,5,6,7,8,9,10,11,12,13,14,15);
#endif
      // B 32x16 frags: j<8 -> k = kk + 8*hi + j ; j>=8 -> +16
      h8 b0 = *(const h8*)&BsT[bc][kk + 8 * hi];
      h8 b1 = *(const h8*)&BsT[bc][kk + 16 + 8 * hi];
      bf0 = __builtin_shufflevector(b0, b1, 0,1,2,3,4,5,6,7,8,9,10,11,12,13,14,15);
      h8 b2 = *(const h8*)&BsT[bc + 16][kk + 8 * hi];
      h8 b3 = *(const h8*)&BsT[bc + 16][kk + 16 + 8 * hi];
      bf1 = __builtin_shufflevector(b2, b3, 0,1,2,3,4,5,6,7,8,9,10,11,12,13,14,15);
      acc0 = __builtin_amdgcn_wmma_f32_16x16x32_f16(
          false, af, false, bf0, (short)0, acc0, false, false);
      acc1 = __builtin_amdgcn_wmma_f32_16x16x32_f16(
          false, af, false, bf1, (short)0, acc1, false, false);
    }
    __syncthreads();
  }

  // C/D 16x16 fp32 layout: VGPR j -> row (j + 8*hi), col lo
#pragma unroll
  for (int j = 0; j < 8; ++j) {
    int r  = bm + wm * 16 + j + 8 * hi;
    int c0 = bn + wn * 32 + lo;
    int c1 = c0 + 16;
    if (r < M) {
      if (c0 < N) {
        float v = acc0[j];
        if (bias) v += bias[c0];
        if (act)  v = tanhf(v);
        C[(size_t)r * ldc + c0] = v;
      }
      if (c1 < N) {
        float v = acc1[j];
        if (bias) v += bias[c1];
        if (act)  v = tanhf(v);
        C[(size_t)r * ldc + c1] = v;
      }
    }
  }
}

// ============================================================
// small helpers
// ============================================================
__global__ void fill_f32(float* p, size_t n) {
  size_t i = (size_t)blockIdx.x * blockDim.x + threadIdx.x;
  if (i < n) p[i] = 0.0f;
}
__global__ void copy_f32(float* dst, const float* src, size_t n) {
  size_t i = (size_t)blockIdx.x * blockDim.x + threadIdx.x;
  if (i < n) dst[i] = src[i];
}

// y[m] = dot(X[m,:], w) + (*bias)    one wave per row
__global__ __launch_bounds__(256) void matvec_f32(
    const float* __restrict__ X, const float* __restrict__ w,
    const float* __restrict__ bias, float* __restrict__ y, int M, int K)
{
  int row  = (int)((blockIdx.x * 256 + threadIdx.x) >> 5);
  int lane = threadIdx.x & 31;
  if (row >= M) return;
  const float* x = X + (size_t)row * K;
  float acc = 0.f;
  for (int j = lane; j < K; j += 32) acc += x[j] * w[j];
  for (int off = 16; off > 0; off >>= 1) acc += __shfl_xor(acc, off, 32);
  if (lane == 0) y[row] = acc + (bias ? *bias : 0.0f);
}

// ============================================================
// GAT edge kernels (one wave per edge, D=128)
// ============================================================
__device__ __forceinline__ unsigned int f32_ord(float f) {
  unsigned int b = __float_as_uint(f);
  return (b & 0x80000000u) ? ~b : (b | 0x80000000u);
}
__device__ __forceinline__ float ord_f32(unsigned int u) {
  unsigned int b = (u & 0x80000000u) ? (u & 0x7fffffffu) : ~u;
  return __uint_as_float(b);
}

// pass1: e = leaky_relu([z_src,z_dst]@a), segment-max into mo[dst]
__global__ __launch_bounds__(256) void edge_pass1(
    const float* __restrict__ z, const float* __restrict__ a,
    const int* __restrict__ src, const int* __restrict__ dst,
    int E, float* __restrict__ e, unsigned int* __restrict__ mo)
{
  int w    = (int)((blockIdx.x * 256 + threadIdx.x) >> 5);
  int lane = threadIdx.x & 31;
  if (w >= E) return;
  int s = src[w], d = dst[w];
  const float* zs = z + (size_t)s * DD;
  const float* zd = z + (size_t)d * DD;
  float acc = 0.f;
#pragma unroll
  for (int t = 0; t < 4; ++t) {
    int j = lane + 32 * t;
    acc += zs[j] * a[j] + zd[j] * a[DD + j];
  }
  for (int off = 16; off > 0; off >>= 1) acc += __shfl_xor(acc, off, 32);
  if (lane == 0) {
    float le = (acc > 0.f) ? acc : 0.01f * acc;  // leaky_relu(0.01)
    e[w] = le;
    atomicMax(mo + d, f32_ord(le));
  }
}

// pass2: w = exp(e - m[dst]); den[dst]+=w; num[dst,:] += w*z[src,:]
__global__ __launch_bounds__(256) void edge_pass2(
    const float* __restrict__ z,
    const int* __restrict__ src, const int* __restrict__ dst,
    int E, const float* __restrict__ e, const unsigned int* __restrict__ mo,
    float* __restrict__ den, float* __restrict__ num)
{
  int w    = (int)((blockIdx.x * 256 + threadIdx.x) >> 5);
  int lane = threadIdx.x & 31;
  if (w >= E) return;
  int s = src[w], d = dst[w];
  float wgt = expf(e[w] - ord_f32(mo[d]));
  if (lane == 0) atomicAdd(den + d, wgt);
  const float* zs = z + (size_t)s * DD;
  float* nd = num + (size_t)d * DD;
#pragma unroll
  for (int t = 0; t < 4; ++t) {
    int j = lane + 32 * t;
    atomicAdd(nd + j, wgt * zs[j]);
  }
}

// out = den>0 ? num/max(den,1e-12) : 0
__global__ void seg_norm(const float* __restrict__ num,
                         const float* __restrict__ den,
                         float* __restrict__ out, int Nn)
{
  size_t i = (size_t)blockIdx.x * blockDim.x + threadIdx.x;
  if (i >= (size_t)Nn * DD) return;
  int n = (int)(i >> 7);
  float dv = den[n];
  out[i] = (dv > 0.f) ? num[i] / fmaxf(dv, 1e-12f) : 0.0f;
}

// kn_new = softmax([sA,sB]) blend of A_,B_   -> write into ek[ITEM:]
__global__ void combine_kn(const float* __restrict__ Ab,
                           const float* __restrict__ Bb,
                           const float* __restrict__ sA,
                           const float* __restrict__ sB,
                           float* __restrict__ knout)
{
  size_t i = (size_t)blockIdx.x * blockDim.x + threadIdx.x;
  if (i >= (size_t)KNOW * DD) return;
  int n = (int)(i >> 7);
  float e0 = sA[n], e1 = sB[n];
  float mx = fmaxf(e0, e1);
  float p0 = expf(e0 - mx), p1 = expf(e1 - mx);
  float s0 = p0 / (p0 + p1);
  knout[i] = s0 * Ab[i] + (1.0f - s0) * Bb[i];
}

// ============================================================
// head: masked means hoisted before the kn projection
// ============================================================
__global__ __launch_bounds__(256) void reduce_beba(
    const int* __restrict__ p_rec, const int* __restrict__ p_target,
    const int* __restrict__ a_rec, const float* __restrict__ ex,
    const float* __restrict__ ans_table, float* __restrict__ h)
{
  int b = blockIdx.x;
  int t = threadIdx.x;
  int Mb = p_target[b] + 1;
  float inv = 1.0f / (float)Mb;
  float acc = 0.f;
  if (t < DD) {
    for (int s = 0; s < Mb; ++s)
      acc += ex[(size_t)p_rec[b * SS + s] * DD + t];
    h[(size_t)b * 384 + t] = acc * inv;
  } else {
    int d = t - DD;
    for (int s = 0; s < Mb; ++s)
      acc += ans_table[(size_t)a_rec[b * SS + s] * DD + d];
    h[(size_t)b * 384 + DD + d] = acc * inv;
  }
}

// q[b,k] = sum_{s<Mb} kn_rec[b,s,k]/kn_num[b,s]   (pure HBM sweep)
__global__ __launch_bounds__(256) void reduce_q(
    const float* __restrict__ kn_rec, const float* __restrict__ kn_num,
    const int* __restrict__ p_target, float* __restrict__ q)
{
  int i = blockIdx.x * 256 + threadIdx.x;   // b*KNOW + k
  int b = i >> 10, k = i & (KNOW - 1);
  int Mb = p_target[b] + 1;
  const float* base = kn_rec + (size_t)b * SS * KNOW + k;
  float acc = 0.f;
  for (int s = 0; s < Mb; ++s)
    acc += base[(size_t)s * KNOW] / kn_num[b * SS + s];
  q[i] = acc;
}

// h[:,256:384] /= Mb
__global__ void scale_hbk(float* __restrict__ h, const int* __restrict__ p_target)
{
  int i = blockIdx.x * 256 + threadIdx.x;   // b*128 + d
  if (i >= BB * DD) return;
  int b = i >> 7, d = i & 127;
  h[(size_t)b * 384 + 256 + d] *= 1.0f / (float)(p_target[b] + 1);
}

// ============================================================
// host-side orchestration
// ============================================================
static inline void launch_gemm(const float* A, const float* B, const float* bias,
                               float* C, int M, int N, int K, int ldc, int act,
                               hipStream_t st)
{
  dim3 g((N + 63) / 64, (M + 63) / 64);
  gemm_wmma_f16<<<g, 256, 0, st>>>(A, B, bias, C, M, N, K, ldc, act);
}
static inline void launch_fill(float* p, size_t n, hipStream_t st) {
  fill_f32<<<(unsigned)((n + 255) / 256), 256, 0, st>>>(p, n);
}
static inline void launch_copy(float* d, const float* s, size_t n, hipStream_t st) {
  copy_f32<<<(unsigned)((n + 255) / 256), 256, 0, st>>>(d, s, n);
}

extern "C" void kernel_launch(void* const* d_in, const int* in_sizes, int n_in,
                              void* d_out, int out_size, void* d_ws, size_t ws_size,
                              hipStream_t stream)
{
  (void)in_sizes; (void)n_in; (void)out_size; (void)ws_size;
  const int*   p_rec     = (const int*)  d_in[0];
  const int*   p_target  = (const int*)  d_in[1];
  const int*   a_rec     = (const int*)  d_in[2];
  const float* kn_rec    = (const float*)d_in[3];
  const float* kn_num    = (const float*)d_in[4];
  const int*   src1      = (const int*)  d_in[5];
  const int*   dst1      = (const int*)  d_in[6];
  const int*   src2      = (const int*)  d_in[7];
  const int*   dst2      = (const int*)  d_in[8];
  const int*   src3      = (const int*)  d_in[9];
  const int*   dst3      = (const int*)  d_in[10];
  const float* kn_table  = (const float*)d_in[11];
  const float* exer_tab  = (const float*)d_in[12];
  const float* ans_table = (const float*)d_in[13];
  const float* gW        = (const float*)d_in[14];
  const float* gA        = (const float*)d_in[15];
  const float* raW1      = (const float*)d_in[16];
  const float* raB1      = (const float*)d_in[17];
  const float* raW2      = (const float*)d_in[18];
  const float* actW1     = (const float*)d_in[19];
  const float* actB1     = (const float*)d_in[20];
  const float* actW2     = (const float*)d_in[21];
  const float* actB2     = (const float*)d_in[22];
  const float* vW1       = (const float*)d_in[23];
  const float* vB1       = (const float*)d_in[24];
  const float* vW2       = (const float*)d_in[25];
  const float* vB2       = (const float*)d_in[26];

  float* out = (float*)d_out;

  // -------- workspace carve-out (floats) --------
  float* ws = (float*)d_ws;
  size_t o = 0;
  float* ek   = ws + o; o += (size_t)NTOT * DD;   // [ex ; kn]
  float* z    = ws + o; o += (size_t)NTOT * DD;   // projected features
  float* num  = ws + o; o += (size_t)NTOT * DD;   // segment numerators / ra temp
  float* gout = ws + o; o += (size_t)NTOT * DD;   // graph layer output
  float* Abuf = ws + o; o += (size_t)KNOW * DD;   // k_dir
  float* Bbuf = ws + o; o += (size_t)KNOW * DD;   // kfe[ITEM:]
  float* den  = ws + o; o += (size_t)NTOT;
  unsigned int* mo = (unsigned int*)(ws + o); o += (size_t)NTOT;
  float* ebuf = ws + o; o += (size_t)E2N;
  float* sA   = ws + o; o += (size_t)KNOW;
  float* sB   = ws + o; o += (size_t)KNOW;
  float* q    = ws + o; o += (size_t)BB * KNOW;
  float* h    = ws + o; o += (size_t)BB * 384;
  float* t1   = ws + o; o += (size_t)BB * LAT;
  float* vh   = ws + o; o += (size_t)3 * BB * LAT;
  float* Tbuf = num;                              // reuse (free between uses)

  float* knp = ek + (size_t)ITEM * DD;            // kn rows inside ek

  // ek = [exer_table ; kn_table]
  launch_copy(ek, exer_tab, (size_t)ITEM * DD, stream);
  launch_copy(knp, kn_table, (size_t)KNOW * DD, stream);

  const int eblk1 = (E1N * 32) / 256;             // 1 wave / edge
  const int eblk2 = (E2N * 32) / 256;

  for (int l = 0; l < LL; ++l) {
    const float* W0 = gW + (size_t)(l * 3 + 0) * DD * DD;
    const float* W1 = gW + (size_t)(l * 3 + 1) * DD * DD;
    const float* W2 = gW + (size_t)(l * 3 + 2) * DD * DD;
    const float* a0 = gA + (size_t)(l * 3 + 0) * 2 * DD;
    const float* a1 = gA + (size_t)(l * 3 + 1) * 2 * DD;
    const float* a2 = gA + (size_t)(l * 3 + 2) * 2 * DD;

    // ---- graph 1: kn -> k_dir (KNOW nodes, E1 edges) ----
    launch_gemm(knp, W0, nullptr, z, KNOW, DD, DD, DD, 0, stream);
    launch_fill((float*)mo, KNOW, stream);
    launch_fill(den, KNOW, stream);
    launch_fill(num, (size_t)KNOW * DD, stream);
    edge_pass1<<<eblk1, 256, 0, stream>>>(z, a0, src1, dst1, E1N, ebuf, mo);
    edge_pass2<<<eblk1, 256, 0, stream>>>(z, src1, dst1, E1N, ebuf, mo, den, num);
    seg_norm<<<(KNOW * DD + 255) / 256, 256, 0, stream>>>(num, den, Abuf, KNOW);

    // ---- graph 2: ek -> kfe (NTOT nodes, E2 edges) ----
    launch_gemm(ek, W1, nullptr, z, NTOT, DD, DD, DD, 0, stream);
    launch_fill((float*)mo, NTOT, stream);
    launch_fill(den, NTOT, stream);
    launch_fill(num, (size_t)NTOT * DD, stream);
    edge_pass1<<<eblk2, 256, 0, stream>>>(z, a1, src2, dst2, E2N, ebuf, mo);
    edge_pass2<<<eblk2, 256, 0, stream>>>(z, src2, dst2, E2N, ebuf, mo, den, num);
    seg_norm<<<(NTOT * DD + 255) / 256, 256, 0, stream>>>(num, den, gout, NTOT);
    launch_copy(Bbuf, gout + (size_t)ITEM * DD, (size_t)KNOW * DD, stream);

    // ---- graph 3: ek -> efk ----
    launch_gemm(ek, W2, nullptr, z, NTOT, DD, DD, DD, 0, stream);
    launch_fill((float*)mo, NTOT, stream);
    launch_fill(den, NTOT, stream);
    launch_fill(num, (size_t)NTOT * DD, stream);
    edge_pass1<<<eblk2, 256, 0, stream>>>(z, a2, src3, dst3, E2N, ebuf, mo);
    edge_pass2<<<eblk2, 256, 0, stream>>>(z, src3, dst3, E2N, ebuf, mo, den, num);
    seg_norm<<<(NTOT * DD + 255) / 256, 256, 0, stream>>>(num, den, gout, NTOT);

    // ---- ra gating: s = softmax([ra(A), ra(B)]) ----
    launch_gemm(Abuf, raW1 + (size_t)l * DD * DD, raB1 + (size_t)l * DD,
                Tbuf, KNOW, DD, DD, DD, 1, stream);
    matvec_f32<<<(KNOW * 32) / 256, 256, 0, stream>>>(Tbuf, raW2 + (size_t)l * DD,
                                                      nullptr, sA, KNOW, DD);
    launch_gemm(Bbuf, raW1 + (size_t)l * DD * DD, raB1 + (size_t)l * DD,
                Tbuf, KNOW, DD, DD, DD, 1, stream);
    matvec_f32<<<(KNOW * 32) / 256, 256, 0, stream>>>(Tbuf, raW2 + (size_t)l * DD,
                                                      nullptr, sB, KNOW, DD);

    // ---- update ek: kn rows then ex rows ----
    combine_kn<<<(KNOW * DD + 255) / 256, 256, 0, stream>>>(Abuf, Bbuf, sA, sB, knp);
    launch_copy(ek, gout, (size_t)ITEM * DD, stream);   // ex = efk[:ITEM]
  }

  // -------- head --------
  reduce_beba<<<BB, 256, 0, stream>>>(p_rec, p_target, a_rec, ek, ans_table, h);
  reduce_q<<<(BB * KNOW) / 256, 256, 0, stream>>>(kn_rec, kn_num, p_target, q);
  // hbk = q @ kn  (masked mean hoisted before this GEMM)
  launch_gemm(q, knp, nullptr, h + 256, BB, DD, KNOW, 384, 0, stream);
  scale_hbk<<<(BB * DD) / 256, 256, 0, stream>>>(h, p_target);

  // actor: logits = tanh(h@actW1+b) @ actW2 + b2
  launch_gemm(h, actW1, actB1, t1, BB, LAT, 384, LAT, 1, stream);
  launch_gemm(t1, actW2, actB2, out, BB, ITEM, LAT, ITEM, 0, stream);

  // critics: 3 heads
  for (int k = 0; k < 3; ++k) {
    launch_gemm(h, vW1 + (size_t)k * 384 * LAT, vB1 + (size_t)k * LAT,
                vh + (size_t)k * BB * LAT, BB, LAT, 384, LAT, 1, stream);
    matvec_f32<<<(BB * 32) / 256, 256, 0, stream>>>(
        vh + (size_t)k * BB * LAT, vW2 + (size_t)k * LAT, vB2 + k,
        out + (size_t)BB * ITEM + (size_t)k * BB, BB, LAT);
  }
}
